// PointTransformer_Unet_PrimSeg_48095043780756
// MI455X (gfx1250) — matrix operand order repaired and checked by reading, not compiled, not run on hardware
//
#include <hip/hip_runtime.h>
#include <hip/hip_bf16.h>

// ---------------------------------------------------------------------------
// PointTransformer U-Net for MI455X (gfx1250, wave32).
// Dense math runs on V_WMMA_F32_16X16X32_BF16 (bf16 operands, f32 accum).
// Weights are pre-packed into per-lane B-fragment layout so each lane's
// fragment is a single contiguous 32-byte load (2x global_load_b128).
// WMMA kernels are templated on C_IN so the k-loop has a compile-time trip
// count: no guards/branches in the hot loop, no accumulator phi copies, and
// dual accumulator chains let partial s_wait + XDL pipelining work.
// ---------------------------------------------------------------------------

typedef __attribute__((ext_vector_type(16))) __bf16 v16bf;
typedef __attribute__((ext_vector_type(8)))  float  v8f;

#define DEV __device__ __forceinline__

DEV int imin_d(int a, int b) { return a < b ? a : b; }

// A-matrix 16x32 bf16 fragment (ISA 7.12.2): lanes 0-15 => M=lane, K in
// {0..7,16..23}; lanes 16-31 => M=lane-16, K in {8..15,24..31}.
DEV v16bf load_A_frag(const __bf16* A, int lda, int kbase, int lane) {
  int m  = lane & 15;
  int hi = (lane >> 4) & 1;
  const __bf16* row = A + m * lda + kbase;
  v16bf a;
#pragma unroll
  for (int j = 0; j < 4; ++j) {
    int k = hi * 8 + 2 * j;
    a[2 * j]     = row[k];
    a[2 * j + 1] = row[k + 1];
  }
#pragma unroll
  for (int j = 0; j < 4; ++j) {
    int k = 16 + hi * 8 + 2 * j;
    a[8 + 2 * j] = row[k];
    a[9 + 2 * j] = row[k + 1];
  }
  return a;
}

// Packed B fragments: layout [tile][kchunk][lane][16 bf16] -- element e of
// lane L is W[kc*32 + (L>=16?8:0) + (e&7) + (e>=8?16:0)][tile*16 + (L&15)].
DEV const __bf16* Bfrag_ptr(const __bf16* Bp, int nk, int tile, int kc, int lane) {
  return Bp + ((((size_t)tile * nk + kc) * 32 + lane) << 4);
}
DEV v16bf load_B_packed(const __bf16* p) { return *(const v16bf*)p; }

// ---------------------------------------------------------------------------
// Pack W [c_in x c_out] f32 row-major into per-lane WMMA B fragments (bf16).
// ---------------------------------------------------------------------------
__global__ void k_packB(const float* __restrict__ W, int c_in, int c_out,
                        __bf16* __restrict__ out) {
  int nk = c_in >> 5, nt = c_out >> 4;
  int total = nt * nk * 32;
  int idx = blockIdx.x * blockDim.x + threadIdx.x;
  if (idx >= total) return;
  int lane = idx & 31;
  int kc   = (idx >> 5) % nk;
  int tile = (idx >> 5) / nk;
  int n  = lane & 15;
  int hi = (lane >> 4) & 1;
  int col = tile * 16 + n;
  int kb  = kc * 32 + hi * 8;
  __bf16* dst = out + ((size_t)idx << 4);
#pragma unroll 4
  for (int e = 0; e < 16; ++e)
    dst[e] = (__bf16)W[(size_t)(kb + (e & 7) + ((e >> 3) << 4)) * c_out + col];
}

// ---------------------------------------------------------------------------
// enc1: x = cat(p0,x0) [N,6]; LN over 6; @W(6x32) -> feats0 [N,32]  (tiny K)
// ---------------------------------------------------------------------------
__global__ void k_enc1(const float* __restrict__ p0, const float* __restrict__ x0,
                       const float* __restrict__ ng, const float* __restrict__ nb,
                       const float* __restrict__ W, float* __restrict__ f0, int n) {
  int i = blockIdx.x * blockDim.x + threadIdx.x;
  if (i >= n) return;
  float x[6];
  x[0] = p0[3 * i]; x[1] = p0[3 * i + 1]; x[2] = p0[3 * i + 2];
  x[3] = x0[3 * i]; x[4] = x0[3 * i + 1]; x[5] = x0[3 * i + 2];
  float mu = 0.f, sq = 0.f;
#pragma unroll
  for (int j = 0; j < 6; ++j) { mu += x[j]; sq += x[j] * x[j]; }
  mu *= (1.f / 6.f);
  float var = sq * (1.f / 6.f) - mu * mu;
  float rstd = rsqrtf(var + 1e-5f);
  float xn[6];
#pragma unroll
  for (int j = 0; j < 6; ++j) xn[j] = (x[j] - mu) * rstd * ng[j] + nb[j];
#pragma unroll 4
  for (int o = 0; o < 32; ++o) {
    float s = 0.f;
#pragma unroll
    for (int j = 0; j < 6; ++j) s += xn[j] * W[j * 32 + o];
    f0[(size_t)i * 32 + o] = s;
  }
}

// ---------------------------------------------------------------------------
// Furthest point sampling: single block, sequential scan with block argmax.
// ---------------------------------------------------------------------------
__global__ void k_fps(const float* __restrict__ xyz, int n, int m,
                      int* __restrict__ out_idx, float* __restrict__ dmin) {
  __shared__ float s_val[1024];
  __shared__ int   s_idx[1024];
  int t = threadIdx.x;
  for (int i = t; i < n; i += blockDim.x) dmin[i] = 3.4e38f;
  if (t == 0) out_idx[0] = 0;
  __syncthreads();
  int last = 0;
  for (int it = 1; it < m; ++it) {
    float lx = xyz[3 * last], ly = xyz[3 * last + 1], lz = xyz[3 * last + 2];
    float best = -1.f; int bidx = 0;
    for (int i = t; i < n; i += blockDim.x) {
      float dx = xyz[3 * i] - lx, dy = xyz[3 * i + 1] - ly, dz = xyz[3 * i + 2] - lz;
      float d  = dx * dx + dy * dy + dz * dz;
      float dm = dmin[i];
      dm = fminf(dm, d);
      dmin[i] = dm;
      if (dm > best) { best = dm; bidx = i; }
    }
    s_val[t] = best; s_idx[t] = bidx;
    __syncthreads();
    for (int s = blockDim.x >> 1; s > 0; s >>= 1) {
      if (t < s && s_val[t + s] > s_val[t]) { s_val[t] = s_val[t + s]; s_idx[t] = s_idx[t + s]; }
      __syncthreads();
    }
    last = s_idx[0];
    if (t == 0) out_idx[it] = last;
    __syncthreads();
  }
}

__global__ void k_gather3(const float* __restrict__ xyz, const int* __restrict__ idx,
                          int m, float* __restrict__ out) {
  int i = blockIdx.x * blockDim.x + threadIdx.x;
  if (i >= m) return;
  int s = idx[i];
  out[3 * i]     = xyz[3 * s];
  out[3 * i + 1] = xyz[3 * s + 1];
  out[3 * i + 2] = xyz[3 * s + 2];
}

// ---------------------------------------------------------------------------
// kNN (k=16): one 64-thread block per query; per-thread sorted top-16 over
// strided refs, then serial 64-way merge by thread 0.
// ---------------------------------------------------------------------------
__global__ void k_knn16(const float* __restrict__ q, const float* __restrict__ r,
                        int n, int* __restrict__ nbr) {
  __shared__ float sd[64 * 16];
  __shared__ int   si[64 * 16];
  int mi = blockIdx.x;
  int t  = threadIdx.x;
  float qx = q[3 * mi], qy = q[3 * mi + 1], qz = q[3 * mi + 2];
  float bd[16]; int bi[16];
#pragma unroll
  for (int j = 0; j < 16; ++j) { bd[j] = 3.4e38f; bi[j] = 0; }
  for (int i = t; i < n; i += 64) {
    float dx = r[3 * i] - qx, dy = r[3 * i + 1] - qy, dz = r[3 * i + 2] - qz;
    float d  = dx * dx + dy * dy + dz * dz;
    if (d < bd[15]) {
      int j = 15;
      while (j > 0 && bd[j - 1] > d) { bd[j] = bd[j - 1]; bi[j] = bi[j - 1]; --j; }
      bd[j] = d; bi[j] = i;
    }
  }
#pragma unroll
  for (int j = 0; j < 16; ++j) { sd[t * 16 + j] = bd[j]; si[t * 16 + j] = bi[j]; }
  __syncthreads();
  if (t == 0) {
    int hp[64];
    for (int z = 0; z < 64; ++z) hp[z] = 0;
    for (int oj = 0; oj < 16; ++oj) {
      float best = 3.4e38f; int bz = 0;
      for (int z = 0; z < 64; ++z) {
        float v = sd[z * 16 + hp[z]];
        if (v < best) { best = v; bz = z; }
      }
      nbr[mi * 16 + oj] = si[bz * 16 + hp[bz]];
      hp[bz]++;
    }
  }
}

// ---------------------------------------------------------------------------
// Shared WMMA tile core: accumulate over all NK chunks with dual chains.
// Compile-time NK => no guards, no phi copies.
// ---------------------------------------------------------------------------
template <int C_IN>
DEV void wmma_k_accum(const __bf16* sA, int lda, const __bf16* Bp, int tile,
                      int lane, v8f& acc0, v8f& acc1) {
  constexpr int NK = C_IN >> 5;
  if constexpr (NK == 1) {
    v16bf a0 = load_A_frag(sA, lda, 0, lane);
    v16bf b0 = load_B_packed(Bfrag_ptr(Bp, 1, tile, 0, lane));
    acc0 = __builtin_amdgcn_wmma_f32_16x16x32_bf16(false, a0, false, b0,
                                                   (short)0, acc0, false, false);
  } else {
    for (int kc = 0; kc < NK; kc += 2) {
      v16bf a0 = load_A_frag(sA, lda, kc * 32, lane);
      v16bf b0 = load_B_packed(Bfrag_ptr(Bp, NK, tile, kc, lane));
      v16bf a1 = load_A_frag(sA, lda, (kc + 1) * 32, lane);
      v16bf b1 = load_B_packed(Bfrag_ptr(Bp, NK, tile, kc + 1, lane));
      if (kc + 2 < NK)
        __builtin_prefetch(Bfrag_ptr(Bp, NK, tile, kc + 2, lane), 0, 1);
      acc0 = __builtin_amdgcn_wmma_f32_16x16x32_bf16(false, a0, false, b0,
                                                     (short)0, acc0, false, false);
      acc1 = __builtin_amdgcn_wmma_f32_16x16x32_bf16(false, a1, false, b1,
                                                     (short)0, acc1, false, false);
    }
  }
}

// ---------------------------------------------------------------------------
// Encoder grouped block: one block per sampled point. Gather 16 neighbor
// feature rows, LayerNorm them, stage bf16 in LDS, WMMA over c_out tiles,
// max-pool the 16 M-rows of each C tile.
// ---------------------------------------------------------------------------
template <int C_IN>
__global__ void k_enc_group_wmma(const float* __restrict__ fin,
                                 const int* __restrict__ nbr,
                                 const float* __restrict__ ng, const float* __restrict__ nb,
                                 const __bf16* __restrict__ Bp, int c_out,
                                 float* __restrict__ fout) {
  constexpr int lda = C_IN + 16;
  __shared__ float  s_sum[256];
  __shared__ float  s_sq[256];
  __shared__ __bf16 sA[16 * lda];

  int t  = threadIdx.x;       // blockDim = 256
  int r  = t >> 4;            // row 0..15 (neighbor within group)
  int c0 = t & 15;
  int src = nbr[blockIdx.x * 16 + r];
  const float* f = fin + (size_t)src * C_IN;
  float sum = 0.f, sq = 0.f;
#pragma unroll
  for (int c = c0; c < C_IN; c += 16) { float v = f[c]; sum += v; sq += v * v; }
  s_sum[t] = sum; s_sq[t] = sq;
  __syncthreads();
  float ts = 0.f, tq = 0.f;
#pragma unroll
  for (int i = 0; i < 16; ++i) { ts += s_sum[r * 16 + i]; tq += s_sq[r * 16 + i]; }
  float mean = ts * (1.f / C_IN);
  float var  = tq * (1.f / C_IN) - mean * mean;
  float rstd = rsqrtf(var + 1e-5f);
#pragma unroll
  for (int c = c0; c < C_IN; c += 16)
    sA[r * lda + c] = (__bf16)((f[c] - mean) * rstd * ng[c] + nb[c]);
  __syncthreads();

  int wave = t >> 5, lane = t & 31;
  int nwaves = blockDim.x >> 5;
  int ntiles = c_out >> 4;
  for (int tile = wave; tile < ntiles; tile += nwaves) {
    v8f acc0 = {}, acc1 = {};
    wmma_k_accum<C_IN>(sA, lda, Bp, tile, lane, acc0, acc1);
    float mx = acc0[0] + acc1[0];
#pragma unroll
    for (int v = 1; v < 8; ++v) mx = fmaxf(mx, acc0[v] + acc1[v]);
    mx = fmaxf(mx, __shfl_xor(mx, 16, 32));   // combine M 0-7 with M 8-15
    if (lane < 16)
      fout[(size_t)blockIdx.x * c_out + tile * 16 + lane] = mx;
  }
}

// ---------------------------------------------------------------------------
// Generic LN + Linear (+bias, + optional residual add) via WMMA.
// One block = one 16-row tile x all c_out tiles (8 waves).
// ---------------------------------------------------------------------------
template <int C_IN>
__global__ void k_ln_linear_wmma(const float* __restrict__ in, int n,
                                 const float* __restrict__ g, const float* __restrict__ b,
                                 const __bf16* __restrict__ Bp, int c_out,
                                 const float* __restrict__ bias,
                                 const float* __restrict__ add,   // may be null
                                 float* __restrict__ out) {
  constexpr int lda = C_IN + 16;
  __shared__ float  s_sum[256];
  __shared__ float  s_sq[256];
  __shared__ __bf16 sA[16 * lda];

  int t  = threadIdx.x;     // blockDim = 256
  int r  = t >> 4;
  int c0 = t & 15;
  int row0 = blockIdx.x * 16;
  int srow = imin_d(row0 + r, n - 1);
  const float* f = in + (size_t)srow * C_IN;
  float sum = 0.f, sq = 0.f;
#pragma unroll
  for (int c = c0; c < C_IN; c += 16) { float v = f[c]; sum += v; sq += v * v; }
  s_sum[t] = sum; s_sq[t] = sq;
  __syncthreads();
  float ts = 0.f, tq = 0.f;
#pragma unroll
  for (int i = 0; i < 16; ++i) { ts += s_sum[r * 16 + i]; tq += s_sq[r * 16 + i]; }
  float mean = ts * (1.f / C_IN);
  float var  = tq * (1.f / C_IN) - mean * mean;
  float rstd = rsqrtf(var + 1e-5f);
#pragma unroll
  for (int c = c0; c < C_IN; c += 16)
    sA[r * lda + c] = (__bf16)((f[c] - mean) * rstd * g[c] + b[c]);
  __syncthreads();

  int wave = t >> 5, lane = t & 31;
  int nwaves = blockDim.x >> 5;
  int ntiles = c_out >> 4;
  int hi  = (lane >> 4) & 1;
  int nn  = lane & 15;
  for (int tile = wave; tile < ntiles; tile += nwaves) {
    v8f acc0 = {}, acc1 = {};
    wmma_k_accum<C_IN>(sA, lda, Bp, tile, lane, acc0, acc1);
    int col = tile * 16 + nn;
    float bc = bias[col];
#pragma unroll
    for (int v = 0; v < 8; ++v) {
      int row = row0 + v + hi * 8;
      if (row < n) {
        float val = acc0[v] + acc1[v] + bc;
        if (add) val += add[(size_t)row * c_out + col];
        out[(size_t)row * c_out + col] = val;
      }
    }
  }
}

// ---------------------------------------------------------------------------
// 3-NN inverse distance weights: one thread per fine point.
// ---------------------------------------------------------------------------
__global__ void k_three_nn(const float* __restrict__ fine, int nf,
                           const float* __restrict__ coarse, int nc,
                           int* __restrict__ idx3, float* __restrict__ w3) {
  int i = blockIdx.x * blockDim.x + threadIdx.x;
  if (i >= nf) return;
  float qx = fine[3 * i], qy = fine[3 * i + 1], qz = fine[3 * i + 2];
  float d0 = 3.4e38f, d1 = 3.4e38f, d2 = 3.4e38f;
  int i0 = 0, i1 = 0, i2 = 0;
  for (int j = 0; j < nc; ++j) {
    float dx = coarse[3 * j] - qx, dy = coarse[3 * j + 1] - qy, dz = coarse[3 * j + 2] - qz;
    float d  = dx * dx + dy * dy + dz * dz;
    if (d < d0)      { d2 = d1; i2 = i1; d1 = d0; i1 = i0; d0 = d; i0 = j; }
    else if (d < d1) { d2 = d1; i2 = i1; d1 = d; i1 = j; }
    else if (d < d2) { d2 = d; i2 = j; }
  }
  float w0 = 1.f / (sqrtf(fmaxf(d0, 0.f)) + 1e-8f);
  float w1 = 1.f / (sqrtf(fmaxf(d1, 0.f)) + 1e-8f);
  float w2 = 1.f / (sqrtf(fmaxf(d2, 0.f)) + 1e-8f);
  float wsum = w0 + w1 + w2;
  idx3[3 * i] = i0; idx3[3 * i + 1] = i1; idx3[3 * i + 2] = i2;
  w3[3 * i] = w0 / wsum; w3[3 * i + 1] = w1 / wsum; w3[3 * i + 2] = w2 / wsum;
}

__global__ void k_upsample3(const float* __restrict__ proj, const int* __restrict__ idx3,
                            const float* __restrict__ w3, int total, int c,
                            float* __restrict__ up) {
  int i = blockIdx.x * blockDim.x + threadIdx.x;
  if (i >= total) return;
  int row = i / c, ch = i - row * c;
  const int*   id = idx3 + 3 * row;
  const float* w  = w3 + 3 * row;
  up[i] = proj[(size_t)id[0] * c + ch] * w[0] +
          proj[(size_t)id[1] * c + ch] * w[1] +
          proj[(size_t)id[2] * c + ch] * w[2];
}

// ---------------------------------------------------------------------------
// Head layer 1: (x@W1 + b1) * g/sqrt(1+eps) + be, ReLU.  32->32 via WMMA.
// ---------------------------------------------------------------------------
__global__ void k_head1_wmma(const float* __restrict__ x, int n,
                             const __bf16* __restrict__ Bp,   // packed, nk=1, 2 tiles
                             const float* __restrict__ b1, const float* __restrict__ g,
                             const float* __restrict__ be, float* __restrict__ h1) {
  __shared__ __bf16 sA[16 * 48];
  int t = threadIdx.x;   // blockDim = 64 (2 waves, 2 tiles)
  int row0 = blockIdx.x * 16;
  for (int i = t; i < 16 * 32; i += 64) {
    int r = i >> 5, c = i & 31;
    int row = imin_d(row0 + r, n - 1);
    sA[r * 48 + c] = (__bf16)x[(size_t)row * 32 + c];
  }
  __syncthreads();
  int wave = t >> 5, lane = t & 31;
  v16bf a  = load_A_frag(sA, 48, 0, lane);
  v16bf bf = load_B_packed(Bfrag_ptr(Bp, 1, wave, 0, lane));
  v8f acc = {};
  acc = __builtin_amdgcn_wmma_f32_16x16x32_bf16(false, a, false, bf,
                                                (short)0, acc, false, false);
  int hi  = (lane >> 4) & 1;
  int col = wave * 16 + (lane & 15);
  float scale = g[col] * rsqrtf(1.f + 1e-5f);
  float bb = b1[col], bee = be[col];
#pragma unroll
  for (int v = 0; v < 8; ++v) {
    int row = row0 + v + hi * 8;
    if (row < n) {
      float y = (acc[v] + bb) * scale + bee;
      h1[(size_t)row * 32 + col] = fmaxf(y, 0.f);
    }
  }
}

// Head layer 2 (c_out in {32,13,2} -- not 16-multiples, tiny: VALU).
__global__ void k_head2(const float* __restrict__ h1, int n,
                        const float* __restrict__ W2, const float* __restrict__ b2,
                        int c_out, float* __restrict__ out) {
  int i = blockIdx.x * blockDim.x + threadIdx.x;
  if (i >= n) return;
  const float* x = h1 + (size_t)i * 32;
  for (int o = 0; o < c_out; ++o) {
    float s = b2[o];
#pragma unroll 8
    for (int k = 0; k < 32; ++k) s += x[k] * W2[k * c_out + o];
    out[(size_t)i * c_out + o] = s;
  }
}

// ---------------------------------------------------------------------------
// Host-side dispatch over compile-time C_IN
// ---------------------------------------------------------------------------
static void launch_enc_group(int ci, int m, hipStream_t stream,
                             const float* fin, const int* nbr,
                             const float* ng, const float* nb,
                             const __bf16* Bp, int co, float* fout) {
  switch (ci) {
    case 32:  k_enc_group_wmma<32><<<m, 256, 0, stream>>>(fin, nbr, ng, nb, Bp, co, fout); break;
    case 64:  k_enc_group_wmma<64><<<m, 256, 0, stream>>>(fin, nbr, ng, nb, Bp, co, fout); break;
    case 128: k_enc_group_wmma<128><<<m, 256, 0, stream>>>(fin, nbr, ng, nb, Bp, co, fout); break;
    case 256: k_enc_group_wmma<256><<<m, 256, 0, stream>>>(fin, nbr, ng, nb, Bp, co, fout); break;
    default:  k_enc_group_wmma<512><<<m, 256, 0, stream>>>(fin, nbr, ng, nb, Bp, co, fout); break;
  }
}

static void launch_ln_linear(int ci, int nblk, hipStream_t stream,
                             const float* in, int n, const float* g, const float* b,
                             const __bf16* Bp, int co, const float* bias,
                             const float* add, float* out) {
  switch (ci) {
    case 32:  k_ln_linear_wmma<32><<<nblk, 256, 0, stream>>>(in, n, g, b, Bp, co, bias, add, out); break;
    case 64:  k_ln_linear_wmma<64><<<nblk, 256, 0, stream>>>(in, n, g, b, Bp, co, bias, add, out); break;
    case 128: k_ln_linear_wmma<128><<<nblk, 256, 0, stream>>>(in, n, g, b, Bp, co, bias, add, out); break;
    case 256: k_ln_linear_wmma<256><<<nblk, 256, 0, stream>>>(in, n, g, b, Bp, co, bias, add, out); break;
    default:  k_ln_linear_wmma<512><<<nblk, 256, 0, stream>>>(in, n, g, b, Bp, co, bias, add, out); break;
  }
}

// ---------------------------------------------------------------------------
// Host orchestration
// ---------------------------------------------------------------------------
struct HeadP { const float *W1, *b1, *g, *be, *W2, *b2; };

extern "C" void kernel_launch(void* const* d_in, const int* in_sizes, int n_in,
                              void* d_out, int out_size, void* d_ws, size_t ws_size,
                              hipStream_t stream) {
  (void)n_in; (void)out_size; (void)ws_size;
  auto F = [&](int i) { return (const float*)d_in[i]; };

  const float *p0, *x0;
  const float *encW[5], *encNg[5], *encNb[5];
  const float *l1W[4], *l1b[4], *l1c[4], *l1g[4];
  const float *l2W[4], *l2b[4], *l2c[4], *l2g[4];
  HeadP emb, cls, bnd;

  // Disambiguate flattening order: insertion order => d_in[1] is x0 (N*3);
  // jax-pytree sorted order => d_in[1] is params.bnd.W1 (1024).
  bool insertion = (in_sizes[1] == in_sizes[0]);
  if (insertion) {
    p0 = F(0); x0 = F(1);
    int ix = 2;
    for (int s = 0; s < 5; ++s) { encNg[s] = F(ix++); encNb[s] = F(ix++); encW[s] = F(ix++); }
    for (int d = 0; d < 4; ++d) {
      l1g[d] = F(ix++); l1b[d] = F(ix++); l1W[d] = F(ix++); l1c[d] = F(ix++);
      l2g[d] = F(ix++); l2b[d] = F(ix++); l2W[d] = F(ix++); l2c[d] = F(ix++);
    }
    HeadP* hs[3] = {&emb, &cls, &bnd};
    for (int h = 0; h < 3; ++h) {
      hs[h]->W1 = F(ix++); hs[h]->b1 = F(ix++); hs[h]->g = F(ix++);
      hs[h]->be = F(ix++); hs[h]->W2 = F(ix++); hs[h]->b2 = F(ix++);
    }
  } else {
    p0 = F(0);
    int ix = 1;
    HeadP* hs[2] = {&bnd, &cls};
    for (int h = 0; h < 2; ++h) {
      hs[h]->W1 = F(ix++); hs[h]->W2 = F(ix++); hs[h]->b1 = F(ix++);
      hs[h]->b2 = F(ix++); hs[h]->be = F(ix++); hs[h]->g = F(ix++);
    }
    for (int d = 0; d < 4; ++d) {
      l1W[d] = F(ix++); l1b[d] = F(ix++); l1c[d] = F(ix++); l1g[d] = F(ix++);
      l2W[d] = F(ix++); l2b[d] = F(ix++); l2c[d] = F(ix++); l2g[d] = F(ix++);
    }
    emb.W1 = F(ix++); emb.W2 = F(ix++); emb.b1 = F(ix++);
    emb.b2 = F(ix++); emb.be = F(ix++); emb.g = F(ix++);
    for (int s = 0; s < 5; ++s) { encW[s] = F(ix++); encNb[s] = F(ix++); encNg[s] = F(ix++); }
    x0 = F(ix++);
  }

  const int N = in_sizes[0] / 3;          // 16384
  int ns[5];  ns[0] = N;
  for (int s = 1; s < 5; ++s) ns[s] = ns[s - 1] / 4 + 1;   // 4097,1025,257,65
  const int planes[5] = {32, 64, 128, 256, 512};

  // ---- workspace bump allocator ----
  char* wp = (char*)d_ws;
  auto alloc = [&](size_t bytes) -> void* {
    void* p = (void*)wp;
    wp += (bytes + 255) & ~(size_t)255;
    return p;
  };

  __bf16* encWb[5] = {nullptr};
  for (int s = 1; s < 5; ++s) encWb[s] = (__bf16*)alloc((size_t)planes[s - 1] * planes[s] * 2);
  __bf16 *l1Wb[4], *l2Wb[4];
  for (int d = 0; d < 4; ++d) {
    int lev = 3 - d, oc = planes[lev], ic = planes[lev + 1];
    l1Wb[d] = (__bf16*)alloc((size_t)oc * oc * 2);
    l2Wb[d] = (__bf16*)alloc((size_t)ic * oc * 2);
  }
  __bf16* hW1b[3];
  for (int h = 0; h < 3; ++h) hW1b[h] = (__bf16*)alloc(32 * 32 * 2);

  float* feats[5];
  for (int s = 0; s < 5; ++s) feats[s] = (float*)alloc((size_t)ns[s] * planes[s] * 4);
  float* xyzs[5];
  xyzs[0] = (float*)p0;   // read-only use
  for (int s = 1; s < 5; ++s) xyzs[s] = (float*)alloc((size_t)ns[s] * 3 * 4);
  float* dmin   = (float*)alloc((size_t)N * 4);
  int*   idxbuf = (int*)alloc((size_t)ns[1] * 4);
  int*   nbrbuf = (int*)alloc((size_t)ns[1] * 16 * 4);
  int*   idx3   = (int*)alloc((size_t)N * 3 * 4);
  float* w3     = (float*)alloc((size_t)N * 3 * 4);
  float *projb[4], *upb[4], *xcb[4];
  for (int d = 0; d < 4; ++d) {
    int lev = 3 - d, oc = planes[lev];
    projb[d] = (float*)alloc((size_t)ns[lev + 1] * oc * 4);
    upb[d]   = (float*)alloc((size_t)ns[lev] * oc * 4);
    xcb[d]   = (float*)alloc((size_t)ns[lev] * oc * 4);
  }
  float* h1 = (float*)alloc((size_t)N * 32 * 4);

  // ---- pack weights to bf16 B-fragment layout ----
  auto pack = [&](const float* src, __bf16* dst, int ci, int co) {
    int total = (co >> 4) * (ci >> 5) * 32;
    k_packB<<<(total + 255) / 256, 256, 0, stream>>>(src, ci, co, dst);
  };
  for (int s = 1; s < 5; ++s) pack(encW[s], encWb[s], planes[s - 1], planes[s]);
  for (int d = 0; d < 4; ++d) {
    int lev = 3 - d, oc = planes[lev], ic = planes[lev + 1];
    pack(l1W[d], l1Wb[d], oc, oc);
    pack(l2W[d], l2Wb[d], ic, oc);
  }
  pack(emb.W1, hW1b[0], 32, 32);
  pack(cls.W1, hW1b[1], 32, 32);
  pack(bnd.W1, hW1b[2], 32, 32);

  // ---- encoder ----
  k_enc1<<<(N + 127) / 128, 128, 0, stream>>>(p0, x0, encNg[0], encNb[0], encW[0], feats[0], N);

  for (int s = 1; s < 5; ++s) {
    int n = ns[s - 1], m = ns[s];
    int ci = planes[s - 1], co = planes[s];
    k_fps<<<1, 1024, 0, stream>>>(xyzs[s - 1], n, m, idxbuf, dmin);
    k_gather3<<<(m + 127) / 128, 128, 0, stream>>>(xyzs[s - 1], idxbuf, m, xyzs[s]);
    k_knn16<<<m, 64, 0, stream>>>(xyzs[s], xyzs[s - 1], n, nbrbuf);
    launch_enc_group(ci, m, stream, feats[s - 1], nbrbuf,
                     encNg[s], encNb[s], encWb[s], co, feats[s]);
  }

  // ---- decoder ----
  const float* xc = feats[4];
  for (int d = 0; d < 4; ++d) {
    int lev = 3 - d, oc = planes[lev], ic = planes[lev + 1];
    int nf = ns[lev], nc = ns[lev + 1];
    k_three_nn<<<(nf + 127) / 128, 128, 0, stream>>>(xyzs[lev], nf, xyzs[lev + 1], nc, idx3, w3);
    launch_ln_linear(ic, (nc + 15) / 16, stream,
                     xc, nc, l2g[d], l2b[d], l2Wb[d], oc, l2c[d], nullptr, projb[d]);
    int total = nf * oc;
    k_upsample3<<<(total + 255) / 256, 256, 0, stream>>>(projb[d], idx3, w3, total, oc, upb[d]);
    launch_ln_linear(oc, (nf + 15) / 16, stream,
                     feats[lev], nf, l1g[d], l1b[d], l1Wb[d], oc, l1c[d], upb[d], xcb[d]);
    xc = xcb[d];
  }

  // ---- heads ----
  float* out = (float*)d_out;
  float* out_emb = out;
  float* out_cls = out + (size_t)N * 32;
  float* out_bnd = out + (size_t)N * 45;
  const HeadP* hs[3] = {&emb, &cls, &bnd};
  const __bf16* hw[3] = {hW1b[0], hW1b[1], hW1b[2]};
  float* outs[3] = {out_emb, out_cls, out_bnd};
  const int cos[3] = {32, 13, 2};
  for (int h = 0; h < 3; ++h) {
    k_head1_wmma<<<(N + 15) / 16, 64, 0, stream>>>(xc, N, hw[h], hs[h]->b1,
                                                   hs[h]->g, hs[h]->be, h1);
    k_head2<<<(N + 127) / 128, 128, 0, stream>>>(h1, N, hs[h]->W2, hs[h]->b2, cos[h], outs[h]);
  }
}
